// DecoderLayer_84404697301735
// MI455X (gfx1250) — matrix operand order repaired and checked by reading, not compiled, not run on hardware
//
#include <hip/hip_runtime.h>

#define DEVI __device__ __forceinline__

typedef __attribute__((ext_vector_type(16))) __bf16 v16bf;
typedef __attribute__((ext_vector_type(8)))  float  v8f;
typedef __attribute__((ext_vector_type(4)))  unsigned u32x4;
typedef __attribute__((ext_vector_type(8)))  int      i32x8;
typedef __attribute__((ext_vector_type(4)))  int      i32x4;

union FragB { unsigned u[8]; v16bf v; };
union FragA { float    f[8]; v8f   v; };

DEVI unsigned short f2bf(float x) {
  unsigned u = __float_as_uint(x);
  u += 0x7FFFu + ((u >> 16) & 1u);          // round-to-nearest-even
  return (unsigned short)(u >> 16);
}
DEVI unsigned pkbf(float a, float b) {
  return (unsigned)f2bf(a) | ((unsigned)f2bf(b) << 16);
}
DEVI float bf2f(unsigned short h) { return __uint_as_float(((unsigned)h) << 16); }

DEVI v8f wmma_bf16(const v16bf& a, const v16bf& b, const v8f& c) {
  return __builtin_amdgcn_wmma_f32_16x16x32_bf16(false, a, false, b, (short)0, c, false, false);
}

// --------------------------- TDM (Tensor Data Mover) -----------------------
#if defined(__has_builtin)
#  if __has_builtin(__builtin_amdgcn_tensor_load_to_lds)
#    define HAVE_TDM 1
#  else
#    define HAVE_TDM 0
#  endif
#else
#  define HAVE_TDM 0
#endif

DEVI unsigned lds_off(const void* p) {
  // generic pointer into LDS aperture: low 32 bits are the LDS byte offset
  return (unsigned)(reinterpret_cast<unsigned long long>(p));
}

#if HAVE_TDM
// 2D tile DMA: global (f32, row length strideElems) -> LDS, optional row pad.
// tileX = contiguous elements per row, tileY = rows.
DEVI void tdm_load_2d(unsigned ldsByteOff, const void* gptr,
                      unsigned tileX, unsigned tileY,
                      unsigned long long strideElems,
                      unsigned padInterval, unsigned padAmount, unsigned padEnable)
{
  unsigned long long ga = reinterpret_cast<unsigned long long>(gptr);
  u32x4 g0;
  g0[0] = 1u;                                           // count=1, user desc
  g0[1] = ldsByteOff;                                   // lds_addr
  g0[2] = (unsigned)(ga & 0xFFFFFFFFull);               // global_addr lo
  g0[3] = (unsigned)((ga >> 32) & 0x01FFFFFFull) | (2u << 30);  // hi | type=2
  i32x8 g1;
  unsigned e0 = (2u << 16);                             // data_size = 4B
  if (padEnable) e0 |= (1u << 20) | (padInterval << 22) | (padAmount << 25);
  g1[0] = (int)e0;
  g1[1] = (int)((tileX & 0xFFFFu) << 16);               // tensor_dim0 = tileX
  g1[2] = (int)(((tileX >> 16) & 0xFFFFu) | ((tileY & 0xFFFFu) << 16)); // tensor_dim1
  g1[3] = (int)(((tileY >> 16) & 0xFFFFu) | ((tileX & 0xFFFFu) << 16)); // tile_dim0
  g1[4] = (int)(tileY & 0xFFFFu);                       // tile_dim1 (tile_dim2=0)
  g1[5] = (int)(strideElems & 0xFFFFFFFFull);           // tensor_dim0_stride lo
  g1[6] = (int)((strideElems >> 32) & 0xFFFFull);       // stride hi (dim1_stride=0)
  g1[7] = 0;
  i32x4 z4 = {0, 0, 0, 0};
#if __clang_major__ >= 23
  i32x8 z8 = {0, 0, 0, 0, 0, 0, 0, 0};
  __builtin_amdgcn_tensor_load_to_lds(g0, g1, z4, z4, z8, 0);
#else
  __builtin_amdgcn_tensor_load_to_lds(g0, g1, z4, z4, 0);
#endif
}
#endif

// ---------------------------------------------------------------------------
// Generic GEMM: C[M,N] = A[M,K](f32) * B[K,N](f32) + bias (+relu), bf16 WMMA.
// M % 128 == 0, N % 64 == 0, K % 32 == 0. Block 256 thr (8 waves), tile 128x64.
// K-tiles staged f32 into LDS by the TDM (double buffered, padded rows),
// converted to bf16 at fragment-assembly time.
// ---------------------------------------------------------------------------
#define APITCH 34   // 32 + 2 pad dwords  (pad_interval code 4 = 32 dw, amount code 1 = 2 dw)
#define BPITCH 65   // 64 + 1 pad dword   (pad_interval code 5 = 64 dw, amount code 0 = 1 dw)

__global__ __launch_bounds__(256) void gemm_bf16w(
    const float* __restrict__ A, const float* __restrict__ B,
    const float* __restrict__ bias, float* __restrict__ C,
    int K, int lda, int ldb, int ldc, int relu)
{
  __shared__ float As[2][128 * APITCH];
  __shared__ float Bs[2][32 * BPITCH];

  const int tid = threadIdx.x;
  const int bm = blockIdx.y * 128;
  const int bn = blockIdx.x * 64;
  const int wid = tid >> 5, lane = tid & 31;
  const int hl = lane >> 4, ll = lane & 15;
  const int m0 = (wid >> 1) * 32, n0 = (wid & 1) * 32;
  const int KITER = K >> 5;

  FragA acc[2][2];
#pragma unroll
  for (int i = 0; i < 2; i++)
#pragma unroll
    for (int j = 0; j < 2; j++)
#pragma unroll
      for (int r = 0; r < 8; r++) acc[i][j].f[r] = 0.f;

  auto stage = [&](int it) {
    const int k0 = it << 5;
    const int bufi = it & 1;
#if HAVE_TDM
    if (tid < 32) {     // one wave issues the DMA descriptors
      tdm_load_2d(lds_off(&As[bufi][0]), A + (size_t)bm * lda + k0,
                  32, 128, (unsigned long long)lda, 4, 1, 1);
      tdm_load_2d(lds_off(&Bs[bufi][0]), B + (size_t)k0 * ldb + bn,
                  64, 32, (unsigned long long)ldb, 5, 0, 1);
    }
#else
    for (int i = tid; i < 128 * 32; i += 256) {
      int r = i >> 5, kk = i & 31;
      As[bufi][r * APITCH + kk] = A[(size_t)(bm + r) * lda + k0 + kk];
    }
    for (int i = tid; i < 32 * 64; i += 256) {
      int kk = i >> 6, c = i & 63;
      Bs[bufi][kk * BPITCH + c] = B[(size_t)(k0 + kk) * ldb + bn + c];
    }
#endif
  };

  stage(0);
  for (int it = 0; it < KITER; ++it) {
    if (it + 1 < KITER) stage(it + 1);
#if HAVE_TDM
    if (tid < 32) {
      if (it + 1 < KITER) __builtin_amdgcn_s_wait_tensorcnt(2);  // current tile done
      else                __builtin_amdgcn_s_wait_tensorcnt(0);
    }
#endif
    __syncthreads();

    const float* Abuf = &As[it & 1][0];
    const float* Bbuf = &Bs[it & 1][0];

    FragB af[2], bfr[2];
#pragma unroll
    for (int mt = 0; mt < 2; mt++) {
      const float* ar = &Abuf[(m0 + mt * 16 + ll) * APITCH];
#pragma unroll
      for (int v = 0; v < 4; v++) {
        const float2 p = *reinterpret_cast<const float2*>(&ar[hl * 8 + 2 * v]);
        af[mt].u[v] = pkbf(p.x, p.y);
      }
#pragma unroll
      for (int v = 0; v < 4; v++) {
        const float2 p = *reinterpret_cast<const float2*>(&ar[16 + hl * 8 + 2 * v]);
        af[mt].u[4 + v] = pkbf(p.x, p.y);
      }
    }
#pragma unroll
    for (int nt = 0; nt < 2; nt++) {
      int c = n0 + nt * 16 + ll;
#pragma unroll
      for (int v = 0; v < 8; v++) {
        int k = (v >> 2) * 16 + hl * 8 + 2 * (v & 3);
        bfr[nt].u[v] = pkbf(Bbuf[k * BPITCH + c], Bbuf[(k + 1) * BPITCH + c]);
      }
    }
#pragma unroll
    for (int mt = 0; mt < 2; mt++)
#pragma unroll
      for (int nt = 0; nt < 2; nt++)
        acc[mt][nt].v = wmma_bf16(af[mt].v, bfr[nt].v, acc[mt][nt].v);
    __syncthreads();
  }

  float bvals[2];
#pragma unroll
  for (int nt = 0; nt < 2; nt++) bvals[nt] = bias[bn + n0 + nt * 16 + ll];

#pragma unroll
  for (int mt = 0; mt < 2; mt++)
#pragma unroll
    for (int nt = 0; nt < 2; nt++)
#pragma unroll
      for (int r = 0; r < 8; r++) {
        int row = bm + m0 + mt * 16 + r + 8 * hl;
        int col = bn + n0 + nt * 16 + ll;
        float v = acc[mt][nt].f[r] + bvals[nt];
        if (relu) v = fmaxf(v, 0.f);
        C[(size_t)row * ldc + col] = v;
      }
}

// ---------------------------------------------------------------------------
// out[row] = LayerNorm(a[row] + resid[row]) * g + b  ; D = 1024, block 256
// ---------------------------------------------------------------------------
__global__ __launch_bounds__(256) void add_ln_kernel(
    const float* __restrict__ a, const float* __restrict__ resid,
    const float* __restrict__ g, const float* __restrict__ bb,
    float* __restrict__ out)
{
  __shared__ float red[256];
  const int D = 1024;
  const int tid = threadIdx.x;
  const size_t base = (size_t)blockIdx.x * D;
  float x[4]; float s = 0.f;
#pragma unroll
  for (int i = 0; i < 4; i++) {
    int c = tid + i * 256;
    x[i] = a[base + c] + resid[base + c];
    s += x[i];
  }
  red[tid] = s; __syncthreads();
  for (int o = 128; o > 0; o >>= 1) { if (tid < o) red[tid] += red[tid + o]; __syncthreads(); }
  float mean = red[0] * (1.0f / D);
  __syncthreads();
  float vs = 0.f;
#pragma unroll
  for (int i = 0; i < 4; i++) { float d = x[i] - mean; vs += d * d; }
  red[tid] = vs; __syncthreads();
  for (int o = 128; o > 0; o >>= 1) { if (tid < o) red[tid] += red[tid + o]; __syncthreads(); }
  float rstd = rsqrtf(red[0] * (1.0f / D) + 1e-5f);
#pragma unroll
  for (int i = 0; i < 4; i++) {
    int c = tid + i * 256;
    out[base + c] = (x[i] - mean) * rstd * g[c] + bb[c];
  }
}

// ---------------------------------------------------------------------------
// qpos[b] = within-sentence index (reference: repeat(offsets)+arange)
// ---------------------------------------------------------------------------
__global__ void qpos_kernel(const int* __restrict__ lens, int nseq, int B,
                            int* __restrict__ qpos)
{
  if (threadIdx.x == 0 && blockIdx.x == 0) {
    int cum = 0, t = 0;
    for (int s = 0; s < nseq; s++) {
      int len = lens[s];
      cum += len;
      int off = len - cum;
      for (int k = 0; k < len && t < B; k++) { qpos[t] = off + t; t++; }
    }
    for (; t < B; t++) qpos[t] = t;
  }
}

// ---------------------------------------------------------------------------
// Pack pos[257,64] f32 -> bf16 pair-packed B layout [kp=32][c=272] (zero pad)
// ---------------------------------------------------------------------------
#define POSW 272
__global__ __launch_bounds__(256) void pos_pack_kernel(
    const float* __restrict__ pos, unsigned* __restrict__ out)
{
  int idx = blockIdx.x * 256 + threadIdx.x;
  if (idx >= 32 * POSW) return;
  int kp = idx / POSW, c = idx % POSW;
  unsigned v = 0;
  if (c < 257) v = pkbf(pos[c * 64 + 2 * kp], pos[c * 64 + 2 * kp + 1]);
  out[idx] = v;
}

// ---------------------------------------------------------------------------
// Flash-style attention with relative-position logits.
// Block = 128 thr (4 waves), handles one (b,h) x 64 query rows.
// Qp[64 x 257] = Qblock @ posT computed in LDS via WMMA; scores streamed in
// 64-col K/V tiles, online softmax, PV accumulated in WMMA fragments.
// posmode 0: c = clip(j - i); posmode 1: c = clip(j - qpos[b]).
// ---------------------------------------------------------------------------
__global__ __launch_bounds__(128) void attn_kernel(
    const float* __restrict__ Q, int ldq, int hsq,
    const float* __restrict__ K, int ldk, int hsk,
    const float* __restrict__ V, int ldv, int hsv,
    const unsigned* __restrict__ posPK,
    const int* __restrict__ qpos,
    float* __restrict__ O, int ldo,
    int Lk, int causal, int posmode, float scale)
{
  extern __shared__ char smem[];
  unsigned*       Qpair = reinterpret_cast<unsigned*>(smem);            // 64*32
  unsigned*       KV    = Qpair + 64 * 32;                              // 32*64
  float*          Ssh   = reinterpret_cast<float*>(KV + 32 * 64);       // 64*66
  float*          mrow  = Ssh + 64 * 66;                                // 64
  float*          lrow  = mrow + 64;                                    // 64
  float*          arow  = lrow + 64;                                    // 64
  float*          red   = arow + 64;                                    // 128
  unsigned short* Qp    = reinterpret_cast<unsigned short*>(red + 128); // 64*POSW

  const int tid = threadIdx.x;
  const int wid = tid >> 5, lane = tid & 31;
  const int hl = lane >> 4, ll = lane & 15;
  const int m0 = wid * 16;
  const int i0 = blockIdx.x * 64;
  const int h  = blockIdx.y;
  const int b  = blockIdx.z;

  const size_t qbase = (size_t)b * 1024 * ldq + (size_t)h * hsq;
  const size_t kbase = (size_t)b * 1024 * ldk + (size_t)h * hsk;
  const size_t vbase = (size_t)b * 1024 * ldv + (size_t)h * hsv;

  // ---- stage Q block (64 x 64) as bf16 k-pairs ----
  for (int i = tid; i < 64 * 32; i += 128) {
    int r = i >> 5, kp = i & 31;
    const float2 q = *reinterpret_cast<const float2*>(Q + qbase + (size_t)(i0 + r) * ldq + 2 * kp);
    Qpair[r * 32 + kp] = pkbf(q.x, q.y);
  }
  if (tid < 64) { mrow[tid] = -3.0e38f; lrow[tid] = 0.f; }
  __syncthreads();

  // ---- Qp = Qblock @ posT  (each wave: its 16 rows x 272 cols) ----
  for (int ct = 0; ct < POSW / 16; ct++) {
    FragA acc;
#pragma unroll
    for (int r = 0; r < 8; r++) acc.f[r] = 0.f;
#pragma unroll
    for (int ks = 0; ks < 2; ks++) {
      FragB af, bfr;
      const unsigned* abase = &Qpair[(m0 + ll) * 32 + ks * 16];
#pragma unroll
      for (int v = 0; v < 4; v++) af.u[v] = abase[hl * 4 + v];
#pragma unroll
      for (int v = 0; v < 4; v++) af.u[4 + v] = abase[8 + hl * 4 + v];
      int col = ct * 16 + ll;
#pragma unroll
      for (int v = 0; v < 8; v++)
        bfr.u[v] = posPK[(size_t)(ks * 16 + (v >> 2) * 8 + hl * 4 + (v & 3)) * POSW + col];
      acc.v = wmma_bf16(af.v, bfr.v, acc.v);
    }
#pragma unroll
    for (int r = 0; r < 8; r++)
      Qp[(m0 + r + 8 * hl) * POSW + ct * 16 + ll] = f2bf(acc.f[r]);
  }
  __syncthreads();

  const int qp_b = posmode ? qpos[b] : 0;
  FragA o[4];
#pragma unroll
  for (int d = 0; d < 4; d++)
#pragma unroll
    for (int r = 0; r < 8; r++) o[d].f[r] = 0.f;

  const int jend = causal ? (i0 + 64) : Lk;
  for (int j0 = 0; j0 < jend; j0 += 64) {
    // ---- K tile (bf16 pairs along dh) ----
    for (int i = tid; i < 32 * 64; i += 128) {
      int kp = i & 31, c = i >> 5;
      const float2 q = *reinterpret_cast<const float2*>(K + kbase + (size_t)(j0 + c) * ldk + 2 * kp);
      KV[kp * 64 + c] = pkbf(q.x, q.y);
    }
    __syncthreads();

    // ---- S = Q K^T tile (16 x 64 per wave) + pos + mask -> LDS ----
#pragma unroll
    for (int nt = 0; nt < 4; nt++) {
      FragA acc;
#pragma unroll
      for (int r = 0; r < 8; r++) acc.f[r] = 0.f;
#pragma unroll
      for (int ks = 0; ks < 2; ks++) {
        FragB af, bfr;
        const unsigned* abase = &Qpair[(m0 + ll) * 32 + ks * 16];
#pragma unroll
        for (int v = 0; v < 4; v++) af.u[v] = abase[hl * 4 + v];
#pragma unroll
        for (int v = 0; v < 4; v++) af.u[4 + v] = abase[8 + hl * 4 + v];
        int col = nt * 16 + ll;
#pragma unroll
        for (int v = 0; v < 8; v++) bfr.u[v] = KV[(ks * 16 + (v >> 2) * 8 + hl * 4 + (v & 3)) * 64 + col];
        acc.v = wmma_bf16(af.v, bfr.v, acc.v);
      }
#pragma unroll
      for (int r = 0; r < 8; r++) {
        int m = m0 + r + 8 * hl;
        int iAbs = i0 + m;
        int j = j0 + nt * 16 + ll;
        int rel = j - (posmode ? qp_b : iAbs);
        rel = rel < -128 ? -128 : (rel > 128 ? 128 : rel);
        float pl = bf2f(Qp[m * POSW + rel + 128]);
        float val = (acc.f[r] + pl) * scale;
        if (causal && j > iAbs) val = -3.0e38f;
        Ssh[m * 66 + nt * 16 + ll] = val;
      }
    }
    __syncthreads();

    // ---- V tile (overwrite KV; pairs along j) ----
    for (int i = tid; i < 32 * 64; i += 128) {
      int d = i & 63, jp = i >> 6;
      float v0 = V[vbase + (size_t)(j0 + 2 * jp) * ldv + d];
      float v1 = V[vbase + (size_t)(j0 + 2 * jp + 1) * ldv + d];
      KV[jp * 64 + d] = pkbf(v0, v1);
    }

    // ---- online softmax over this 64-col block ----
    {
      int r2 = tid >> 1, half = tid & 1;
      float* srow = &Ssh[r2 * 66 + half * 32];
      float pmax = -3.0e38f;
#pragma unroll
      for (int k = 0; k < 32; k++) pmax = fmaxf(pmax, srow[k]);
      red[tid] = pmax;
      __syncthreads();
      if (tid < 64) {
        float nm = fmaxf(mrow[tid], fmaxf(red[2 * tid], red[2 * tid + 1]));
        arow[tid] = __expf(mrow[tid] - nm);
        mrow[tid] = nm;
      }
      __syncthreads();
      float mr = mrow[r2];
      float ps = 0.f;
#pragma unroll
      for (int k = 0; k < 32; k++) { float e = __expf(srow[k] - mr); srow[k] = e; ps += e; }
      red[tid] = ps;
      __syncthreads();
      if (tid < 64) lrow[tid] = lrow[tid] * arow[tid] + red[2 * tid] + red[2 * tid + 1];
      __syncthreads();
    }

    // ---- rescale O fragments, accumulate P @ V ----
#pragma unroll
    for (int r = 0; r < 8; r++) {
      float al = arow[m0 + r + 8 * hl];
#pragma unroll
      for (int d = 0; d < 4; d++) o[d].f[r] *= al;
    }
#pragma unroll
    for (int dt = 0; dt < 4; dt++) {
#pragma unroll
      for (int ks = 0; ks < 2; ks++) {
        FragB af, bfr;
        const float* prow = &Ssh[(m0 + ll) * 66 + ks * 32];
#pragma unroll
        for (int v = 0; v < 4; v++) {
          const float2 p = *reinterpret_cast<const float2*>(&prow[hl * 8 + 2 * v]);
          af.u[v] = pkbf(p.x, p.y);
        }
#pragma unroll
        for (int v = 0; v < 4; v++) {
          const float2 p = *reinterpret_cast<const float2*>(&prow[16 + hl * 8 + 2 * v]);
          af.u[4 + v] = pkbf(p.x, p.y);
        }
        int col = dt * 16 + ll;
#pragma unroll
        for (int v = 0; v < 8; v++) bfr.u[v] = KV[(ks * 16 + (v >> 2) * 8 + hl * 4 + (v & 3)) * 64 + col];
        o[dt].v = wmma_bf16(af.v, bfr.v, o[dt].v);
      }
    }
    __syncthreads();
  }

  // ---- epilogue: O /= l, merged-head store ----
#pragma unroll
  for (int r = 0; r < 8; r++) {
    int m = m0 + r + 8 * hl;
    float linv = 1.0f / lrow[m];
    size_t orow = (size_t)(b * 1024 + i0 + m) * ldo + h * 64;
#pragma unroll
    for (int dt = 0; dt < 4; dt++)
      O[orow + dt * 16 + ll] = o[dt].f[r] * linv;
  }
}

// ---------------------------------------------------------------------------
// Host orchestration
// ---------------------------------------------------------------------------
static inline void launch_gemm(const float* A, const float* B, const float* bias,
                               float* C, int M, int N, int K,
                               int lda, int ldb, int ldc, int relu, hipStream_t s)
{
  dim3 grid(N / 64, M / 128);
  gemm_bf16w<<<grid, 256, 0, s>>>(A, B, bias, C, K, lda, ldb, ldc, relu);
}

extern "C" void kernel_launch(void* const* d_in, const int* in_sizes, int n_in,
                              void* d_out, int out_size, void* d_ws, size_t ws_size,
                              hipStream_t stream) {
  (void)n_in; (void)out_size; (void)ws_size;
  const float* char_enc   = (const float*)d_in[0];
  const float* word_enc   = (const float*)d_in[1];
  const float* self_input = (const float*)d_in[2];
  const int*   sent_lens  = (const int*)d_in[6];
  const float* qkv_w = (const float*)d_in[7];  const float* qkv_b = (const float*)d_in[8];
  const float* o1_w  = (const float*)d_in[9];  const float* o1_b  = (const float*)d_in[10];
  const float* pos1  = (const float*)d_in[11];
  const float* q2_w  = (const float*)d_in[12]; const float* q2_b  = (const float*)d_in[13];
  const float* k2_w  = (const float*)d_in[14]; const float* k2_b  = (const float*)d_in[15];
  const float* v2_w  = (const float*)d_in[16]; const float* v2_b  = (const float*)d_in[17];
  const float* o2_w  = (const float*)d_in[18]; const float* o2_b  = (const float*)d_in[19];
  const float* pos2  = (const float*)d_in[20];
  const float* q3_w  = (const float*)d_in[21]; const float* q3_b  = (const float*)d_in[22];
  const float* k3_w  = (const float*)d_in[23]; const float* k3_b  = (const float*)d_in[24];
  const float* v3_w  = (const float*)d_in[25]; const float* v3_b  = (const float*)d_in[26];
  const float* o3_w  = (const float*)d_in[27]; const float* o3_b  = (const float*)d_in[28];
  const float* pos3  = (const float*)d_in[29];
  const float* f1_w  = (const float*)d_in[30]; const float* f1_b  = (const float*)d_in[31];
  const float* f2_w  = (const float*)d_in[32]; const float* f2_b  = (const float*)d_in[33];
  const float* ln1_g = (const float*)d_in[34]; const float* ln1_b = (const float*)d_in[35];
  const float* ln2_g = (const float*)d_in[36]; const float* ln2_b = (const float*)d_in[37];
  const float* ln3_g = (const float*)d_in[38]; const float* ln3_b = (const float*)d_in[39];

  const int Bb = 4, L = 1024, D = 1024, H = 16, F = 4096;
  const int T = Bb * L;                 // 4096 token rows
  const float scale = 0.125f;           // 1/sqrt(64)

  float* ws = (float*)d_ws;
  size_t off = 0;
  float* QKV   = ws + off; off += (size_t)T * 3 * D;   // also Q/K/V for layers 2/3
  float* Qb    = QKV;
  float* Kb    = QKV + (size_t)T * D;
  float* Vb    = QKV + (size_t)2 * T * D;
  float* attnO = ws + off; off += (size_t)T * D;
  float* proj  = ws + off; off += (size_t)T * D;
  float* xA    = ws + off; off += (size_t)T * D;
  float* xB    = ws + off; off += (size_t)T * D;
  float* hid   = ws + off; off += (size_t)T * F;
  unsigned* pp1 = (unsigned*)(ws + off); off += 32 * POSW;
  unsigned* pp2 = (unsigned*)(ws + off); off += 32 * POSW;
  unsigned* pp3 = (unsigned*)(ws + off); off += 32 * POSW;
  int* qposb    = (int*)(ws + off);      off += 4;

  const size_t attnSmem =
      (64 * 32 + 32 * 64) * sizeof(unsigned) + (64 * 66 + 64 * 3 + 128) * sizeof(float)
      + 64 * POSW * sizeof(unsigned short);
  const dim3 agrid(L / 64, H, Bb);

  // ---- prep ----
  qpos_kernel<<<1, 32, 0, stream>>>(sent_lens, in_sizes[6], Bb, qposb);
  const int ppBlocks = (32 * POSW + 255) / 256;
  pos_pack_kernel<<<ppBlocks, 256, 0, stream>>>(pos1, pp1);
  pos_pack_kernel<<<ppBlocks, 256, 0, stream>>>(pos2, pp2);
  pos_pack_kernel<<<ppBlocks, 256, 0, stream>>>(pos3, pp3);

  // ---- sublayer 1: fused QKV proj, causal rel-pos attention ----
  launch_gemm(self_input, qkv_w, qkv_b, QKV, T, 3 * D, D, D, 3 * D, 3 * D, 0, stream);
  attn_kernel<<<agrid, 128, attnSmem, stream>>>(
      QKV,        3 * D, 192,   // Q
      QKV + 64,   3 * D, 192,   // K
      QKV + 128,  3 * D, 192,   // V
      pp1, qposb, attnO, D, L, /*causal=*/1, /*posmode=*/0, scale);
  launch_gemm(attnO, o1_w, o1_b, proj, T, D, D, D, D, D, 0, stream);
  add_ln_kernel<<<T, 256, 0, stream>>>(proj, self_input, ln1_g, ln1_b, xA);

  // ---- sublayer 2: cross-attention to char encoder ----
  launch_gemm(xA,       q2_w, q2_b, Qb, T, D, D, D, D, D, 0, stream);
  launch_gemm(char_enc, k2_w, k2_b, Kb, T, D, D, D, D, D, 0, stream);
  launch_gemm(char_enc, v2_w, v2_b, Vb, T, D, D, D, D, D, 0, stream);
  attn_kernel<<<agrid, 128, attnSmem, stream>>>(
      Qb, D, 64, Kb, D, 64, Vb, D, 64,
      pp2, qposb, attnO, D, L, 0, 0, scale);
  launch_gemm(attnO, o2_w, o2_b, proj, T, D, D, D, D, D, 0, stream);
  add_ln_kernel<<<T, 256, 0, stream>>>(proj, xA, ln2_g, ln2_b, xB);

  // ---- sublayer 3: cross-attention to word encoder (per-batch q positions) --
  launch_gemm(xB,       q3_w, q3_b, Qb, T, D, D, D, D, D, 0, stream);
  launch_gemm(word_enc, k3_w, k3_b, Kb, T, D, D, D, D, D, 0, stream);
  launch_gemm(word_enc, v3_w, v3_b, Vb, T, D, D, D, D, D, 0, stream);
  attn_kernel<<<agrid, 128, attnSmem, stream>>>(
      Qb, D, 64, Kb, D, 64, Vb, D, 64,
      pp3, qposb, attnO, D, L, 0, /*posmode=*/1, scale);
  launch_gemm(attnO, o3_w, o3_b, proj, T, D, D, D, D, D, 0, stream);
  add_ln_kernel<<<T, 256, 0, stream>>>(proj, xB, ln3_g, ln3_b, xA);

  // ---- FFN (reference reuses ln3 at output) ----
  launch_gemm(xA,  f1_w, f1_b, hid,  T, F, D, D, F, F, /*relu=*/1, stream);
  launch_gemm(hid, f2_w, f2_b, proj, T, D, F, F, D, D, 0, stream);
  add_ln_kernel<<<T, 256, 0, stream>>>(proj, xA, ln3_g, ln3_b, (float*)d_out);
}
